// CausalEVAttention_60146722013350
// MI455X (gfx1250) — matrix operand order repaired and checked by reading, not compiled, not run on hardware
//
#include <hip/hip_runtime.h>

// ---------------------------------------------------------------------------
// CausalEVAttention for MI455X (gfx1250, wave32, WMMA).
// bf16 data / f32 accumulate via v_wmma_f32_16x16x32_bf16 for all matmuls;
// fp32 VALU for tiny per-chunk RFA statistics. GEMM k-loop is two-stage
// manually unrolled (ping/pong register sets, no rotation moves) so fragment
// loads overlap the WMMA pipe with partial s_wait_loadcnt.
// ---------------------------------------------------------------------------

typedef __bf16 bf16_t;
typedef __attribute__((ext_vector_type(8)))  __bf16 v8bf;
typedef __attribute__((ext_vector_type(16))) __bf16 v16bf;
typedef __attribute__((ext_vector_type(8)))  float  v8f;

#define E_   1024
#define H_   16
#define N_   4096
#define B_   4
#define D_   64
#define WSZ  128
#define EXT_ 128
#define G_   (N_ / WSZ)     // 32 windows
#define NC_  (N_ / 128)     // 32 chunks
#define MROWS (N_ * B_)     // 16384 GEMM rows
#define NEGINF (-1.0e9f)

// ---- WMMA fragment loaders (CDNA5 16-bit layouts, wave32) -----------------
// A (16x32, row-major source): lane L holds row M=L%16.
//   lanes 0-15 : elems[0..7]=K k0+0..7,  elems[8..15]=K k0+16..23
//   lanes 16-31: elems[0..7]=K k0+8..15, elems[8..15]=K k0+24..31
__device__ __forceinline__ v16bf ldA(const bf16_t* p, int ld, int k0, int lane) {
  const bf16_t* s = p + (lane & 15) * ld + k0 + ((lane >> 4) << 3);
  v8bf lo = *(const v8bf*)(s);
  v8bf hi = *(const v8bf*)(s + 16);
  v16bf f;
#pragma unroll
  for (int i = 0; i < 8; ++i) { f[i] = lo[i]; f[i + 8] = hi[i]; }
  return f;
}

// B (32x16): source stored column-major-as-rows: Bt[n][k] contiguous in k.
// lane L holds col N=L%16; lanes 0-15 K=k0+0..7,16..23, lanes 16-31 K=k0+8..15,24..31.
__device__ __forceinline__ v16bf ldB(const bf16_t* p, int ld, int k0, int lane) {
  const bf16_t* s = p + (lane & 15) * ld + k0 + ((lane >> 4) << 3);
  v8bf lo = *(const v8bf*)(s);
  v8bf hi = *(const v8bf*)(s + 16);
  v16bf f;
#pragma unroll
  for (int i = 0; i < 8; ++i) { f[i] = lo[i]; f[i + 8] = hi[i]; }
  return f;
}

__device__ __forceinline__ v8f wmma_bf16(v16bf a, v16bf b, v8f c) {
  return __builtin_amdgcn_wmma_f32_16x16x32_bf16(false, a, false, b,
                                                 (short)0, c, false, false);
}

__device__ __forceinline__ v8f vzero8() {
  v8f z = {0.f, 0.f, 0.f, 0.f, 0.f, 0.f, 0.f, 0.f};
  return z;
}

// ---- Kernel 1: fp32 -> bf16 ----------------------------------------------
__global__ void k_cvt_bf16(const float* __restrict__ src,
                           bf16_t* __restrict__ dst, int n) {
  int i = blockIdx.x * blockDim.x + threadIdx.x;
  int stride = gridDim.x * blockDim.x;
  for (; i < n; i += stride) dst[i] = (bf16_t)src[i];
}

// ---- Kernel 2/5: WMMA GEMM  C(M=16384,N=1024) = A * W^T + bias -----------
// Each wave computes a 32x64 tile (8 accumulators). k-loop advances 64 per
// iteration with ping/pong fragment register sets: stage-P WMMAs run while
// stage-Q fragments load, and vice versa -> no rotation moves, partial waits.
// mode 0: scatter bf16 into (b,h,n,d) layout with scale (QKV projections)
// mode 1: fp32 row-major store (output projection -> d_out)
__global__ __launch_bounds__(128)
void k_gemm(const bf16_t* __restrict__ A, const bf16_t* __restrict__ Bt,
            const float* __restrict__ bias, float scale, int mode,
            bf16_t* __restrict__ dst_bf, float* __restrict__ dst_f) {
  const int K = E_;
  int lane = threadIdx.x & 31;
  int wave = threadIdx.x >> 5;                  // 0..3
  int rowBase = blockIdx.y * 128 + wave * 32;   // 32 rows per wave
  int colBase = blockIdx.x * 64;
  const bf16_t* A0 = A + (size_t)rowBase * K;
  const bf16_t* A1 = A0 + (size_t)16 * K;
  const bf16_t* B0 = Bt + (size_t)(colBase +  0) * K;
  const bf16_t* B1 = Bt + (size_t)(colBase + 16) * K;
  const bf16_t* B2 = Bt + (size_t)(colBase + 32) * K;
  const bf16_t* B3 = Bt + (size_t)(colBase + 48) * K;

  v8f acc[8];
#pragma unroll
  for (int t = 0; t < 8; ++t) acc[t] = vzero8();

  // stage P: fragments for k0; stage Q: fragments for k0+32
  v16bf pa0 = ldA(A0, K, 0, lane);
  v16bf pa1 = ldA(A1, K, 0, lane);
  v16bf pb0 = ldB(B0, K, 0, lane);
  v16bf pb1 = ldB(B1, K, 0, lane);
  v16bf pb2 = ldB(B2, K, 0, lane);
  v16bf pb3 = ldB(B3, K, 0, lane);

  for (int k0 = 0; k0 < K; k0 += 64) {
    // issue stage-Q loads (k0+32), then compute stage P (k0)
    v16bf qa0 = ldA(A0, K, k0 + 32, lane);
    v16bf qa1 = ldA(A1, K, k0 + 32, lane);
    v16bf qb0 = ldB(B0, K, k0 + 32, lane);
    v16bf qb1 = ldB(B1, K, k0 + 32, lane);
    v16bf qb2 = ldB(B2, K, k0 + 32, lane);
    v16bf qb3 = ldB(B3, K, k0 + 32, lane);
    __builtin_prefetch(A0 + k0 + 256, 0, 3);    // near-scope L2 pull-ahead
    __builtin_prefetch(A1 + k0 + 256, 0, 3);

    acc[0] = wmma_bf16(pa0, pb0, acc[0]);
    acc[1] = wmma_bf16(pa0, pb1, acc[1]);
    acc[2] = wmma_bf16(pa0, pb2, acc[2]);
    acc[3] = wmma_bf16(pa0, pb3, acc[3]);
    acc[4] = wmma_bf16(pa1, pb0, acc[4]);
    acc[5] = wmma_bf16(pa1, pb1, acc[5]);
    acc[6] = wmma_bf16(pa1, pb2, acc[6]);
    acc[7] = wmma_bf16(pa1, pb3, acc[7]);

    // issue stage-P loads for next iteration (k0+64), compute stage Q (k0+32)
    int k2 = k0 + 64;
    if (k2 < K) {
      pa0 = ldA(A0, K, k2, lane);
      pa1 = ldA(A1, K, k2, lane);
      pb0 = ldB(B0, K, k2, lane);
      pb1 = ldB(B1, K, k2, lane);
      pb2 = ldB(B2, K, k2, lane);
      pb3 = ldB(B3, K, k2, lane);
    }
    acc[0] = wmma_bf16(qa0, qb0, acc[0]);
    acc[1] = wmma_bf16(qa0, qb1, acc[1]);
    acc[2] = wmma_bf16(qa0, qb2, acc[2]);
    acc[3] = wmma_bf16(qa0, qb3, acc[3]);
    acc[4] = wmma_bf16(qa1, qb0, acc[4]);
    acc[5] = wmma_bf16(qa1, qb1, acc[5]);
    acc[6] = wmma_bf16(qa1, qb2, acc[6]);
    acc[7] = wmma_bf16(qa1, qb3, acc[7]);
  }

  int colLane = lane & 15;
  int rAdd = (lane >> 4) * 8;   // C/D: lanes 16-31 hold rows M+8
#pragma unroll
  for (int mi = 0; mi < 2; ++mi) {
#pragma unroll
    for (int t = 0; t < 4; ++t) {
      int col = colBase + 16 * t + colLane;
      float bv = bias[col];
#pragma unroll
      for (int r = 0; r < 8; ++r) {
        int row = rowBase + mi * 16 + r + rAdd; // row = n*B + b
        float val = (acc[mi * 4 + t][r] + bv) * scale;
        if (mode == 0) {
          int n = row >> 2, bq = row & 3;
          int h = col >> 6, d = col & 63;
          dst_bf[(((size_t)bq * H_ + h) * N_ + n) * D_ + d] = (bf16_t)val;
        } else {
          dst_f[(size_t)row * E_ + col] = val;
        }
      }
    }
  }
}

// ---- Kernel 3: per-(b,h,chunk) RFA statistics -----------------------------
// Produces rf_k_bar (bf16) and rfa_chunk (bf16), both (b,h,c,d).
__global__ __launch_bounds__(128)
void k_chunk(const bf16_t* __restrict__ Q, const bf16_t* __restrict__ Kb,
             const bf16_t* __restrict__ Vb,
             const float* __restrict__ qw, const float* __restrict__ qb_,
             const float* __restrict__ qg, const float* __restrict__ qbe,
             const float* __restrict__ kw, const float* __restrict__ kb_,
             const float* __restrict__ kg, const float* __restrict__ kbe,
             bf16_t* __restrict__ kbar_bf, bf16_t* __restrict__ rchunk_bf) {
  int idx = blockIdx.x;
  int c = idx & (NC_ - 1), bh = idx >> 5;       // bh = b*H + h
  int tid = threadIdx.x;
  const size_t base = ((size_t)bh * N_ + (size_t)c * 128) * D_;
  const float scaling = 0.125f;                 // d^-0.5, d=64

  __shared__ float sq[64], sk[64], qlin[64], klin[64], smu[64], lg[128];

  if (tid < 64) {                               // chunk means (q pre-scaled)
    float aq = 0.f, ak = 0.f;
    for (int p = 0; p < 128; ++p) {
      aq += (float)Q[base + p * D_ + tid];
      ak += (float)Kb[base + p * D_ + tid];
    }
    sq[tid] = aq * (1.f / 128.f);
    sk[tid] = ak * (1.f / 128.f);
  }
  __syncthreads();
  if (tid < 64) {                               // 64x64 linear
    float a = qb_[tid], b2 = kb_[tid];
    for (int j = 0; j < 64; ++j) {
      a  += qw[tid * 64 + j] * sq[j];
      b2 += kw[tid * 64 + j] * sk[j];
    }
    qlin[tid] = a; klin[tid] = b2;
  }
  __syncthreads();
  float qbar = 0.f, kbar = 0.f;
  if (tid < 64) {                               // layernorm
    float mq = 0.f, mk = 0.f;
    for (int j = 0; j < 64; ++j) { mq += qlin[j]; mk += klin[j]; }
    mq *= (1.f / 64.f); mk *= (1.f / 64.f);
    float vq = 0.f, vk = 0.f;
    for (int j = 0; j < 64; ++j) {
      float dq = qlin[j] - mq, dk = klin[j] - mk;
      vq += dq * dq; vk += dk * dk;
    }
    vq *= (1.f / 64.f); vk *= (1.f / 64.f);
    qbar = (qlin[tid] - mq) * rsqrtf(vq + 1e-5f) * qg[tid] + qbe[tid];
    kbar = (klin[tid] - mk) * rsqrtf(vk + 1e-5f) * kg[tid] + kbe[tid];
  }
  __syncthreads();
  if (tid < 64) {
    smu[tid] = qbar + kbar;                     // mu
    kbar_bf[((size_t)bh * NC_ + c) * D_ + tid] = (bf16_t)kbar;
  }
  __syncthreads();
  {                                             // log_proj_k
    float dot = 0.f, k2 = 0.f;
    for (int d = 0; d < 64; ++d) {
      float kv = (float)Kb[base + tid * D_ + d];
      dot += smu[d] * kv;
      k2  += kv * kv;
    }
    lg[tid] = dot * scaling - scaling * k2 * 0.5f;
  }
  __syncthreads();
  float mx = -1e30f;
  for (int p = 0; p < 128; ++p) mx = fmaxf(mx, lg[p]);
  float ssum = 0.f;
  for (int p = 0; p < 128; ++p) ssum += __expf(lg[p] - mx);
  float beta = __expf(lg[tid] - mx) / ssum;
  __syncthreads();
  lg[tid] = beta;
  __syncthreads();
  if (tid < 64) {                               // rfa_chunk = beta^T V
    float rc = 0.f;
    for (int p = 0; p < 128; ++p) rc += lg[p] * (float)Vb[base + p * D_ + tid];
    rchunk_bf[((size_t)bh * NC_ + c) * D_ + tid] = (bf16_t)rc;
  }
}

// ---- Kernel 4: windowed attention per (window g, b*h) ---------------------
#define PROB_LD 288   // 256 local + 32 rfa logits per row
#define VT_LD   264
#define RT_LD   40
__global__ __launch_bounds__(256)
void k_attn(const bf16_t* __restrict__ Q, const bf16_t* __restrict__ Kb,
            const bf16_t* __restrict__ Vb, const bf16_t* __restrict__ kbar_bf,
            const bf16_t* __restrict__ rchunk_bf, bf16_t* __restrict__ outp) {
  int g  = blockIdx.x;                          // window 0..31
  int bh = blockIdx.y;                          // b*H + h
  int b = bh >> 4, h = bh & 15;
  int tid = threadIdx.x;
  int lane = tid & 31, wave = tid >> 5;         // 8 waves x 16 rows

  extern __shared__ __align__(16) bf16_t smem[];
  bf16_t* Vt = smem;                            // [64][VT_LD]  V^T window
  bf16_t* Rt = Vt + 64 * VT_LD;                 // [64][RT_LD]  rfa_chunk^T
  bf16_t* Pr = Rt + 64 * RT_LD;                 // [8][16][PROB_LD] probs

  const size_t qkvBase = (size_t)bh * N_ * D_;
  int kwinStart = g * WSZ - EXT_;

  for (int i = tid; i < 256 * 64; i += 256) {   // stage V^T
    int j = i >> 6, d = i & 63;
    int nv = kwinStart + j;
    float v = (nv >= 0) ? (float)Vb[qkvBase + (size_t)nv * D_ + d] : 0.f;
    Vt[d * VT_LD + j] = (bf16_t)v;
  }
  for (int i = tid; i < 32 * 64; i += 256) {    // stage rfa_chunk^T
    int c = i >> 6, d = i & 63;
    Rt[d * RT_LD + c] = rchunk_bf[((size_t)bh * NC_ + c) * D_ + d];
  }
  __syncthreads();

  const bf16_t* Qrow = Q + qkvBase + (size_t)(g * WSZ + wave * 16) * D_;
  v16bf aq0 = ldA(Qrow, D_, 0, lane);
  v16bf aq1 = ldA(Qrow, D_, 32, lane);

  // B-fragment base for logit tile t (clamped in-bounds; OOB tiles masked)
  auto tileBase = [&](int t) -> const bf16_t* {
    if (t < 16) {
      int rb = kwinStart + t * 16;
      if (rb < 0) rb = 0;                        // g==0, t<8: masked anyway
      return Kb + qkvBase + (size_t)rb * D_;
    }
    return kbar_bf + ((size_t)bh * NC_ + (t - 16) * 16) * D_;
  };

  v8f sacc[18];
#pragma unroll
  for (int t = 0; t < 18; ++t) sacc[t] = vzero8();

  // 18 logit tiles, B-fragments double-buffered across tiles (fully unrolled)
  v16bf cb0 = ldB(tileBase(0), D_, 0, lane);
  v16bf cb1 = ldB(tileBase(0), D_, 32, lane);
#pragma unroll
  for (int t = 0; t < 18; ++t) {
    v16bf nb0 = cb0, nb1 = cb1;
    if (t + 1 < 18) {
      const bf16_t* nb = tileBase(t + 1);
      nb0 = ldB(nb, D_, 0, lane);
      nb1 = ldB(nb, D_, 32, lane);
    }
    sacc[t] = wmma_bf16(aq0, cb0, sacc[t]);
    sacc[t] = wmma_bf16(aq1, cb1, sacc[t]);
    cb0 = nb0; cb1 = nb1;
  }

  int colLane = lane & 15;
  int rAdd = (lane >> 4) * 8;
#pragma unroll
  for (int t = 0; t < 18; ++t) {                // masks
#pragma unroll
    for (int r = 0; r < 8; ++r) {
      int iw = wave * 16 + r + rAdd;            // row in window
      bool ok;
      if (t < 16) {
        int j = t * 16 + colLane;
        ok = (j <= iw + EXT_) && (j >= EXT_ - g * WSZ);
      } else {
        int c = (t - 16) * 16 + colLane;
        ok = (c < g);                           // chunk fully in the past
      }
      if (!ok) sacc[t][r] = NEGINF;
    }
  }

  bf16_t* Pw = Pr + (size_t)wave * 16 * PROB_LD;
#pragma unroll
  for (int r = 0; r < 8; ++r) {                 // joint softmax over 288
    float mx = -1e30f;
#pragma unroll
    for (int t = 0; t < 18; ++t) mx = fmaxf(mx, sacc[t][r]);
#pragma unroll
    for (int off = 8; off >= 1; off >>= 1) mx = fmaxf(mx, __shfl_xor(mx, off, 16));
    float s = 0.f;
#pragma unroll
    for (int t = 0; t < 18; ++t) {
      float e = __expf(sacc[t][r] - mx);
      sacc[t][r] = e; s += e;
    }
#pragma unroll
    for (int off = 8; off >= 1; off >>= 1) s += __shfl_xor(s, off, 16);
    float inv = 1.f / s;
    int rowL = r + rAdd;
#pragma unroll
    for (int t = 0; t < 18; ++t)
      Pw[rowL * PROB_LD + t * 16 + colLane] = (bf16_t)(sacc[t][r] * inv);
  }
  __syncthreads();

  // out = la @ w_v + ga @ rfa_chunk : 9 k-steps (8 local + 1 rfa),
  // A/B fragments double-buffered across steps (fully unrolled).
  v8f oacc[4];
#pragma unroll
  for (int t = 0; t < 4; ++t) oacc[t] = vzero8();

  auto pvA = [&](int s) -> v16bf { return ldA(Pw, PROB_LD, s * 32, lane); };
  auto pvB = [&](int s, int t) -> v16bf {
    if (s < 8) return ldB(Vt + t * 16 * VT_LD, VT_LD, s * 32, lane);
    return ldB(Rt + t * 16 * RT_LD, RT_LD, 0, lane);
  };

  v16bf pa = pvA(0);
  v16bf pb0 = pvB(0, 0), pb1 = pvB(0, 1), pb2 = pvB(0, 2), pb3 = pvB(0, 3);
#pragma unroll
  for (int s = 0; s < 9; ++s) {
    v16bf xa = pa, xb0 = pb0, xb1 = pb1, xb2 = pb2, xb3 = pb3;
    if (s + 1 < 9) {
      pa = pvA(s + 1);
      pb0 = pvB(s + 1, 0); pb1 = pvB(s + 1, 1);
      pb2 = pvB(s + 1, 2); pb3 = pvB(s + 1, 3);
    }
    oacc[0] = wmma_bf16(xa, xb0, oacc[0]);
    oacc[1] = wmma_bf16(xa, xb1, oacc[1]);
    oacc[2] = wmma_bf16(xa, xb2, oacc[2]);
    oacc[3] = wmma_bf16(xa, xb3, oacc[3]);
  }

#pragma unroll
  for (int t = 0; t < 4; ++t) {                 // store as (n, b, e) bf16
#pragma unroll
    for (int r = 0; r < 8; ++r) {
      int n = g * WSZ + wave * 16 + r + rAdd;
      int d = t * 16 + colLane;
      outp[((size_t)n * B_ + b) * E_ + h * D_ + d] = (bf16_t)oacc[t][r];
    }
  }
}

// ---------------------------------------------------------------------------
extern "C" void kernel_launch(void* const* d_in, const int* in_sizes, int n_in,
                              void* d_out, int out_size, void* d_ws, size_t ws_size,
                              hipStream_t stream) {
  const float* query  = (const float*)d_in[0];
  const float* Wq     = (const float*)d_in[1];
  const float* bq     = (const float*)d_in[2];
  const float* Wk     = (const float*)d_in[3];
  const float* bk     = (const float*)d_in[4];
  const float* Wv     = (const float*)d_in[5];
  const float* bv     = (const float*)d_in[6];
  const float* Wo     = (const float*)d_in[7];
  const float* bo     = (const float*)d_in[8];
  const float* mu_q_w = (const float*)d_in[9];
  const float* mu_q_b = (const float*)d_in[10];
  const float* mu_q_g = (const float*)d_in[11];
  const float* mu_q_be= (const float*)d_in[12];
  const float* mu_k_w = (const float*)d_in[13];
  const float* mu_k_b = (const float*)d_in[14];
  const float* mu_k_g = (const float*)d_in[15];
  const float* mu_k_be= (const float*)d_in[16];
  (void)in_sizes; (void)n_in; (void)out_size; (void)ws_size;

  char* ws = (char*)d_ws;
  const size_t szAct  = (size_t)MROWS * E_ * sizeof(bf16_t);     // 32 MB
  const size_t szW    = (size_t)E_ * E_ * sizeof(bf16_t);        // 2 MB
  const size_t szQKV  = (size_t)B_ * H_ * N_ * D_ * sizeof(bf16_t);
  const size_t szBar  = (size_t)B_ * H_ * NC_ * D_ * sizeof(bf16_t);

  bf16_t* qbf   = (bf16_t*)ws;            ws += szAct;
  bf16_t* wqbf  = (bf16_t*)ws;            ws += szW;
  bf16_t* wkbf  = (bf16_t*)ws;            ws += szW;
  bf16_t* wvbf  = (bf16_t*)ws;            ws += szW;
  bf16_t* wobf  = (bf16_t*)ws;            ws += szW;
  bf16_t* Qbf   = (bf16_t*)ws;            ws += szQKV;
  bf16_t* Kbf   = (bf16_t*)ws;            ws += szQKV;
  bf16_t* Vbf   = (bf16_t*)ws;            ws += szQKV;
  bf16_t* kbarb = (bf16_t*)ws;            ws += szBar;
  bf16_t* rchkb = (bf16_t*)ws;            ws += szBar;
  bf16_t* aobf  = (bf16_t*)ws;            ws += szAct;

  // 1) fp32 -> bf16
  k_cvt_bf16<<<2048, 256, 0, stream>>>(query, qbf, MROWS * E_);
  k_cvt_bf16<<<512, 256, 0, stream>>>(Wq, wqbf, E_ * E_);
  k_cvt_bf16<<<512, 256, 0, stream>>>(Wk, wkbf, E_ * E_);
  k_cvt_bf16<<<512, 256, 0, stream>>>(Wv, wvbf, E_ * E_);
  k_cvt_bf16<<<512, 256, 0, stream>>>(Wo, wobf, E_ * E_);

  // 2) QKV projections (scatter to (b,h,n,d); Q pre-scaled by d^-0.5)
  dim3 ggrid(E_ / 64, MROWS / 128);
  k_gemm<<<ggrid, 128, 0, stream>>>(qbf, wqbf, bq, 0.125f, 0, Qbf, nullptr);
  k_gemm<<<ggrid, 128, 0, stream>>>(qbf, wkbf, bk, 1.0f,   0, Kbf, nullptr);
  k_gemm<<<ggrid, 128, 0, stream>>>(qbf, wvbf, bv, 1.0f,   0, Vbf, nullptr);

  // 3) RFA chunk statistics
  k_chunk<<<B_ * H_ * NC_, 128, 0, stream>>>(Qbf, Kbf, Vbf,
      mu_q_w, mu_q_b, mu_q_g, mu_q_be, mu_k_w, mu_k_b, mu_k_g, mu_k_be,
      kbarb, rchkb);

  // 4) windowed attention
  dim3 agrid(G_, B_ * H_);
  size_t shbytes = (size_t)(64 * VT_LD + 64 * RT_LD + 8 * 16 * PROB_LD) * sizeof(bf16_t);
  k_attn<<<agrid, 256, shbytes, stream>>>(Qbf, Kbf, Vbf, kbarb, rchkb, aobf);

  // 5) output projection -> fp32 (N, B, E)
  k_gemm<<<ggrid, 128, 0, stream>>>(aobf, wobf, bo, 1.0f, 1, nullptr, (float*)d_out);
}
